// ConvolutionLayer_38208029065639
// MI455X (gfx1250) — matrix operand order repaired and checked by reading, not compiled, not run on hardware
//
#include <hip/hip_runtime.h>

// ---------------------------------------------------------------------------
// Implicit-GEMM 3x3 valid conv via split-precision BF16 WMMA (gfx1250).
//   x: [32,128,112,112] f32   w: [256,128,3,3] f32  -> out: [32,256,110,110] f32
// GEMM: M=Cout=256, K=Cin*9=1152 (k = (kh*3+kw)*128 + cin), N = one output row.
// Each f32 is split into bf16 hi + bf16 lo; D += Ah*Bh + Ah*Bl + Al*Bh (f32 acc).
// Wave tiling: 8 waves/WG, wave owns M-tiles {w, w+8} x 7 N-tiles (112 cols).
// A fragments: prepacked fragment-order bf16 in d_ws, software-pipelined +
// prefetched 2 K-steps ahead. B fragments: LDS bf16 planes, ds_load_b128.
// Output: non-temporal stores (write-once streaming, preserve L2 for A/x).
// ---------------------------------------------------------------------------

typedef __attribute__((ext_vector_type(16))) __bf16 v16bf;
typedef __attribute__((ext_vector_type(8)))  float  v8f;

#define CIN   128
#define COUT  256
#define HIN   112
#define WIN   112
#define OHOUT 110
#define OWOUT 110
#define KTOT  1152         // CIN * 9
#define NKS   36           // KTOT / 32
#define CPAD  136          // padded cin dim (ushorts) -> 68 dwords, bank-friendly
#define XROWS 3
#define XCOLS 114          // 112 + 2 (kw shift)
#define XS_ELEMS (XROWS * XCOLS * CPAD)        // 46512 ushorts per (hi|lo) plane
#define APACK_HALF 294912                      // 16*36*2*32*8 ushorts per (hi|lo)

__device__ __forceinline__ void bf16_split(float w, unsigned short& hi, unsigned short& lo) {
    unsigned u  = __float_as_uint(w);
    unsigned th = u + 0x7fffu + ((u >> 16) & 1u);          // RNE to bf16
    hi = (unsigned short)(th >> 16);
    float hf = __uint_as_float((unsigned)hi << 16);
    float l  = w - hf;
    unsigned ul = __float_as_uint(l);
    unsigned tl = ul + 0x7fffu + ((ul >> 16) & 1u);
    lo = (unsigned short)(tl >> 16);
}

// ---------------------------------------------------------------------------
// Pre-pass: rearrange weights into WMMA A-fragment order (bf16 hi & lo planes).
// Flat layout (ushorts): [sel(hi/lo)][mt 0..15][ks 0..35][h 0..1][lane 0..31][j 0..7]
//   lane<16 : row = mt*16+lane,     K = ks*32 + h*16 + j        (V0..3 / V4..7)
//   lane>=16: row = mt*16+lane-16,  K = ks*32 + h*16 + 8 + j
// matching the ISA "16-bit A-Matrix 16x32" VGPR layout exactly.
// ---------------------------------------------------------------------------
__global__ void __launch_bounds__(256) conv_prepack_weights(
        const float* __restrict__ w, unsigned short* __restrict__ apack) {
    int t = blockIdx.x * blockDim.x + threadIdx.x;     // 36864 = 16*36*2*32
    if (t >= 16 * 36 * 2 * 32) return;
    int lane = t & 31;
    int h    = (t >> 5) & 1;
    int mk   = t >> 6;                 // mt*36 + ks
    int ks   = mk % 36;
    int mt   = mk / 36;
    int row  = mt * 16 + (lane & 15);
    int kbase = ks * 32 + h * 16 + ((lane >> 4) * 8);

    unsigned short hi[8], lo[8];
#pragma unroll
    for (int j = 0; j < 8; ++j) {
        int k   = kbase + j;
        int p   = k >> 7;              // 0..8  (kh*3+kw)
        int cin = k & 127;
        float val = w[(size_t)row * KTOT + cin * 9 + p];
        bf16_split(val, hi[j], lo[j]);
    }
    uint4 qh, ql;
    qh.x = (unsigned)hi[0] | ((unsigned)hi[1] << 16);
    qh.y = (unsigned)hi[2] | ((unsigned)hi[3] << 16);
    qh.z = (unsigned)hi[4] | ((unsigned)hi[5] << 16);
    qh.w = (unsigned)hi[6] | ((unsigned)hi[7] << 16);
    ql.x = (unsigned)lo[0] | ((unsigned)lo[1] << 16);
    ql.y = (unsigned)lo[2] | ((unsigned)lo[3] << 16);
    ql.z = (unsigned)lo[4] | ((unsigned)lo[5] << 16);
    ql.w = (unsigned)lo[6] | ((unsigned)lo[7] << 16);
    *(uint4*)(apack + (size_t)t * 8)              = qh;
    *(uint4*)(apack + APACK_HALF + (size_t)t * 8) = ql;
}

// A-fragment offset (ushorts) for (mt, ks, lane): 1KB contiguous per (mt,ks).
__device__ __forceinline__ size_t a_off(int mt, int ks, int lane) {
    return (size_t)(mt * NKS + ks) * 512 + (size_t)lane * 8;
}

__device__ __forceinline__ v16bf loadA(const unsigned short* base, size_t o) {
    union { uint4 q[2]; v16bf v; } u;
    u.q[0] = *(const uint4*)(base + o);          // h=0 half (K 0..7 / 8..15)
    u.q[1] = *(const uint4*)(base + o + 256);    // h=1 half (K 16..23 / 24..31)
    return u.v;
}

// ---------------------------------------------------------------------------
// Main kernel: one workgroup (8 waves) per (b, oh). Wave w owns M-tiles {w, w+8}
// and all 7 N-tiles (112 cols, masked to 110). K loop = 36 steps of 32.
// ---------------------------------------------------------------------------
__global__ void __launch_bounds__(256) conv_wmma_bf16x3(
        const float* __restrict__ x,
        const unsigned short* __restrict__ apack,
        float* __restrict__ out) {
    extern __shared__ __align__(16) unsigned short xs[];
    unsigned short* xh = xs;                  // [XROWS][XCOLS][CPAD] hi plane
    unsigned short* xl = xs + XS_ELEMS;       // lo plane

    const int oh  = blockIdx.x;
    const int b   = blockIdx.y;
    const int tid = threadIdx.x;

    // ---- Stage x rows oh..oh+2 into LDS as bf16 hi/lo, [r][col][cin] ----
    // Vectorized: float4 loads (28 chunks of 4 cols per (cin,r)).
    const float* xb = x + (size_t)b * CIN * (HIN * WIN);
    for (int idx = tid; idx < CIN * XROWS * (WIN / 4); idx += 256) {
        int cin = idx / (XROWS * (WIN / 4));
        int rem = idx - cin * (XROWS * (WIN / 4));
        int r   = rem / (WIN / 4);
        int c4  = rem - r * (WIN / 4);
        float4 v = *(const float4*)(xb + (size_t)cin * (HIN * WIN) + (oh + r) * WIN + c4 * 4);
        float vv[4] = {v.x, v.y, v.z, v.w};
#pragma unroll
        for (int j = 0; j < 4; ++j) {
            unsigned short h16, l16;
            bf16_split(vv[j], h16, l16);
            int o = (r * XCOLS + (c4 * 4 + j)) * CPAD + cin;
            xh[o] = h16;
            xl[o] = l16;
        }
    }
    __syncthreads();

    const int lane  = tid & 31;
    const int wave  = tid >> 5;
    const int lhalf = lane >> 4;      // 0 or 1
    const int l15   = lane & 15;

    v8f acc[2][7];
#pragma unroll
    for (int mi = 0; mi < 2; ++mi)
#pragma unroll
        for (int nt = 0; nt < 7; ++nt)
            acc[mi][nt] = (v8f){0.f, 0.f, 0.f, 0.f, 0.f, 0.f, 0.f, 0.f};

    const unsigned short* Ah = apack;
    const unsigned short* Al = apack + APACK_HALF;
    const int mt0 = wave, mt1 = wave + 8;

    // ---- Software pipeline: preload A fragments for ks=0 ----
    v16bf ah0 = loadA(Ah, a_off(mt0, 0, lane));
    v16bf al0 = loadA(Al, a_off(mt0, 0, lane));
    v16bf ah1 = loadA(Ah, a_off(mt1, 0, lane));
    v16bf al1 = loadA(Al, a_off(mt1, 0, lane));

    for (int ks = 0; ks < NKS; ++ks) {
        // Preload next K-step's A fragments (clamped, branchless).
        int ksn = (ks + 1 < NKS) ? (ks + 1) : (NKS - 1);
        v16bf nah0 = loadA(Ah, a_off(mt0, ksn, lane));
        v16bf nal0 = loadA(Al, a_off(mt0, ksn, lane));
        v16bf nah1 = loadA(Ah, a_off(mt1, ksn, lane));
        v16bf nal1 = loadA(Al, a_off(mt1, ksn, lane));

        // Prefetch A blocks two K-steps ahead (1KB per (sel,mt); 32B/lane).
        {
            int ksp = (ks + 2 < NKS) ? (ks + 2) : (NKS - 1);
            size_t p0 = (size_t)(mt0 * NKS + ksp) * 512 + (size_t)lane * 16;
            size_t p1 = (size_t)(mt1 * NKS + ksp) * 512 + (size_t)lane * 16;
            __builtin_prefetch(Ah + p0, 0, 3);
            __builtin_prefetch(Al + p0, 0, 3);
            __builtin_prefetch(Ah + p1, 0, 3);
            __builtin_prefetch(Al + p1, 0, 3);
        }

        int p  = ks >> 2;             // 0..8
        int kh = p / 3;
        int kw = p - kh * 3;
        int cinbase = (ks & 3) * 32 + lhalf * 8;    // B: lanes 0-15 K=0..7,16..23; 16-31 K=8..15,24..31

        const unsigned short* xrh = xh + (kh * XCOLS) * CPAD;
        const unsigned short* xrl = xl + (kh * XCOLS) * CPAD;

#pragma unroll
        for (int nt = 0; nt < 7; ++nt) {
            int col = nt * 16 + l15 + kw;
            int o   = col * CPAD + cinbase;           // 16B-aligned (272B col stride)
            union F { uint4 q[2]; v16bf v; } bh, bl;
            bh.q[0] = *(const uint4*)(xrh + o);
            bh.q[1] = *(const uint4*)(xrh + o + 16);
            bl.q[0] = *(const uint4*)(xrl + o);
            bl.q[1] = *(const uint4*)(xrl + o + 16);

            acc[0][nt] = __builtin_amdgcn_wmma_f32_16x16x32_bf16(
                false, ah0, false, bh.v, (short)0, acc[0][nt], false, false);
            acc[0][nt] = __builtin_amdgcn_wmma_f32_16x16x32_bf16(
                false, ah0, false, bl.v, (short)0, acc[0][nt], false, false);
            acc[0][nt] = __builtin_amdgcn_wmma_f32_16x16x32_bf16(
                false, al0, false, bh.v, (short)0, acc[0][nt], false, false);

            acc[1][nt] = __builtin_amdgcn_wmma_f32_16x16x32_bf16(
                false, ah1, false, bh.v, (short)0, acc[1][nt], false, false);
            acc[1][nt] = __builtin_amdgcn_wmma_f32_16x16x32_bf16(
                false, ah1, false, bl.v, (short)0, acc[1][nt], false, false);
            acc[1][nt] = __builtin_amdgcn_wmma_f32_16x16x32_bf16(
                false, al1, false, bh.v, (short)0, acc[1][nt], false, false);
        }

        ah0 = nah0; al0 = nal0; ah1 = nah1; al1 = nal1;
    }

    // ---- Store: C layout — VGPR r: lanes0-15 M=r, lanes16-31 M=r+8; N=lane%16 ----
    // Non-temporal: output is write-once streaming; keep L2 for A-pack / x rows.
    float* ob = out + ((size_t)b * COUT * OHOUT + oh) * OWOUT;
#pragma unroll
    for (int mi = 0; mi < 2; ++mi) {
        int mtile = (mi == 0) ? mt0 : mt1;
#pragma unroll
        for (int nt = 0; nt < 7; ++nt) {
            int n = nt * 16 + l15;
            if (n < OWOUT) {
#pragma unroll
                for (int r = 0; r < 8; ++r) {
                    int m = mtile * 16 + r + lhalf * 8;
                    __builtin_nontemporal_store(acc[mi][nt][r],
                        ob + (size_t)m * (OHOUT * OWOUT) + n);
                }
            }
        }
    }
}

extern "C" void kernel_launch(void* const* d_in, const int* in_sizes, int n_in,
                              void* d_out, int out_size, void* d_ws, size_t ws_size,
                              hipStream_t stream) {
    (void)in_sizes; (void)n_in; (void)out_size; (void)ws_size;
    const float* x = (const float*)d_in[0];
    const float* w = (const float*)d_in[1];
    float* out = (float*)d_out;
    unsigned short* apack = (unsigned short*)d_ws;   // needs 2*294912*2 = 1,179,648 B

    // Pre-pass: 36864 threads rearrange + bf16-split the weights.
    conv_prepack_weights<<<144, 256, 0, stream>>>(w, apack);

    // Main conv: one workgroup per (oh, b); 186,048 B dynamic LDS (< 320 KB/WGP).
    dim3 grid(OHOUT, 32);
    size_t smem = (size_t)2 * XS_ELEMS * sizeof(unsigned short);
    conv_wmma_bf16x3<<<grid, 256, smem, stream>>>(x, apack, out);
}
